// SelfAttention_60790967108213
// MI455X (gfx1250) — matrix operand order repaired and checked by reading, not compiled, not run on hardware
//
#include <hip/hip_runtime.h>

typedef _Float16 half_t;
typedef __attribute__((ext_vector_type(16))) _Float16 v16h;
typedef __attribute__((ext_vector_type(8)))  _Float16 v8h;
typedef __attribute__((ext_vector_type(8)))  float    v8f;
typedef __attribute__((ext_vector_type(4)))  unsigned int v4u;
typedef __attribute__((ext_vector_type(8)))  int      v8i;
typedef __attribute__((ext_vector_type(4)))  int      v4i;

static constexpr int BATCH = 4;
static constexpr int C     = 256;
static constexpr int N     = 4096;   // 64*64
static constexpr int DQK   = 32;
static constexpr int VPAD  = 40;     // LDS row stride (halves) for V tile: 32 data + 8 pad

#define WMMA_F16(a, b, c) \
  __builtin_amdgcn_wmma_f32_16x16x32_f16(false, (a), false, (b), (short)0, (c), false, false)

// ---------------------------------------------------------------------------
// Projection: out[n][d] = sum_c W[d][c] * x[c][n] + bias[d], stored (B, N, D=32) f16
// ---------------------------------------------------------------------------
__global__ __launch_bounds__(32) void proj_qk_kernel(
    const float* __restrict__ x, const float* __restrict__ W,
    const float* __restrict__ bias, half_t* __restrict__ out)
{
  const int lane = threadIdx.x & 31;
  const int lo = lane & 15, hi = lane >> 4;
  const int nbase = blockIdx.x * 64;
  const int dbase = blockIdx.y * 16;
  const int b     = blockIdx.z;
  const float* xb = x + (size_t)b * C * N;
  half_t*      ob = out + (size_t)b * N * DQK;

  v8f acc[4] = {};
  for (int cb = 0; cb < C; cb += 32) {
    v16h afrag;
    const float* wr = W + (size_t)(dbase + lo) * C + cb;
#pragma unroll
    for (int h = 0; h < 8; ++h) {
      afrag[h]     = (half_t)wr[16 * hi + h];
      afrag[h + 8] = (half_t)wr[16 * hi + 8 + h];
    }
#pragma unroll
    for (int s = 0; s < 4; ++s) {
      const int n = nbase + s * 16 + lo;
      v16h bfrag;
#pragma unroll
      for (int h = 0; h < 16; ++h)
        bfrag[h] = (half_t)xb[(size_t)(cb + 16 * hi + h) * N + n];
      acc[s] = WMMA_F16(afrag, bfrag, acc[s]);
    }
  }
  float bval[8];
#pragma unroll
  for (int r = 0; r < 8; ++r) bval[r] = bias[dbase + 8 * hi + r];
#pragma unroll
  for (int s = 0; s < 4; ++s) {
    const int n = nbase + s * 16 + lo;
    v8h pk;
#pragma unroll
    for (int r = 0; r < 8; ++r) pk[r] = (half_t)(acc[s][r] + bval[r]);
    *(v8h*)(ob + (size_t)n * DQK + dbase + 8 * hi) = pk;
  }
}

// ---------------------------------------------------------------------------
// V projection: v[c][n] = sum_cc Wv[c][cc] * x[cc][n] + bv[c], stored (B, C, N) f16
// ---------------------------------------------------------------------------
__global__ __launch_bounds__(32) void proj_v_kernel(
    const float* __restrict__ x, const float* __restrict__ W,
    const float* __restrict__ bias, half_t* __restrict__ out)
{
  const int lane = threadIdx.x & 31;
  const int lo = lane & 15, hi = lane >> 4;
  const int nbase = blockIdx.x * 64;
  const int cbase = blockIdx.y * 16;
  const int b     = blockIdx.z;
  const float* xb = x + (size_t)b * C * N;
  half_t*      ob = out + (size_t)b * C * N;

  v8f acc[4] = {};
  for (int cb = 0; cb < C; cb += 32) {
    v16h afrag;
    const float* wr = W + (size_t)(cbase + lo) * C + cb;
#pragma unroll
    for (int h = 0; h < 8; ++h) {
      afrag[h]     = (half_t)wr[16 * hi + h];
      afrag[h + 8] = (half_t)wr[16 * hi + 8 + h];
    }
#pragma unroll
    for (int s = 0; s < 4; ++s) {
      const int n = nbase + s * 16 + lo;
      v16h bfrag;
#pragma unroll
      for (int h = 0; h < 16; ++h)
        bfrag[h] = (half_t)xb[(size_t)(cb + 16 * hi + h) * N + n];
      acc[s] = WMMA_F16(afrag, bfrag, acc[s]);
    }
  }
#pragma unroll
  for (int s = 0; s < 4; ++s) {
    const int n = nbase + s * 16 + lo;
#pragma unroll
    for (int r = 0; r < 8; ++r) {
      const int c = cbase + r + 8 * hi;
      ob[(size_t)c * N + n] = (half_t)(acc[s][r] + bias[c]);
    }
  }
}

// ---------------------------------------------------------------------------
// TDM: stage a 32(kv) x 256(c) f16 tile of V (row stride N) into LDS with a
// 40-half padded row stride. 2D descriptor, data_size=2B,
// pad_interval=16 dwords (code 3), pad_amount=4 dwords (code 3).
// This toolchain's builtin takes 6 operands (g0, g1, g2, g3, g4, cpol).
// ---------------------------------------------------------------------------
static __device__ inline void tdm_load_v(const half_t* gsrc, unsigned int lds_off)
{
  const unsigned long long ga = (unsigned long long)(uintptr_t)gsrc;
  v4u g0;
  g0[0] = 1u;                                   // count=1, user descriptor
  g0[1] = lds_off;                              // LDS byte address
  g0[2] = (unsigned int)ga;                     // global_addr[31:0]
  g0[3] = (unsigned int)(ga >> 32) | (2u << 30);// global_addr[56:32] | type=2
  v8i g1;
  g1[0] = (int)((1u << 16)        // data_size = 2 bytes
              | (1u << 20)        // pad_enable
              | (3u << 22)        // pad_interval: 16 dwords
              | (3u << 25));      // pad_amount: 4 dwords
  g1[1] = (int)(4096u << 16);     // tensor_dim0 = 4096 (low 16 bits << 16)
  g1[2] = (int)(256u << 16);      // tensor_dim0 hi=0 | tensor_dim1 = 256
  g1[3] = (int)(32u << 16);       // tensor_dim1 hi=0 | tile_dim0 = 32 (kv)
  g1[4] = 256;                    // tile_dim1 = 256 (c), tile_dim2 = 0
  g1[5] = 4096;                   // tensor_dim0_stride = N
  g1[6] = 0;                      // stride hi | tensor_dim1_stride lo = 0
  g1[7] = 0;
  v4i g2 = {};                    // 2D tensor: groups 2/3 unused
  v4i g3 = {};
  v8i g4 = {};                    // extra operand on this toolchain (unused)
  __builtin_amdgcn_tensor_load_to_lds(g0, g1, g2, g3, g4, 0);
}

// ---------------------------------------------------------------------------
// Flash attention + epilogue. One wave: 16 query rows x all 256 channels.
// V tiles staged block-wide in LDS by the Tensor Data Mover (double buffered).
// Per 32-kv step: 2 WMMA (S = Q K^T), online softmax, P relayout via LDS,
// 16 WMMA (O += P V^T). Epilogue: out = gamma * O/l + x.
// ---------------------------------------------------------------------------
__global__ __launch_bounds__(256) void attn_kernel(
    const float*  __restrict__ x,
    const half_t* __restrict__ qt,   // (B, N, 32)
    const half_t* __restrict__ kt,   // (B, N, 32)
    const half_t* __restrict__ vm,   // (B, C, N)
    const float*  __restrict__ gamma_p,
    float* __restrict__ out)
{
  __shared__ alignas(128) _Float16 vtile[2][C * VPAD];   // 2 x 20KB, TDM destination
  __shared__ alignas(64)  _Float16 pbuf[8][16][48];      // per-wave P relayout slab

  const int wv   = threadIdx.x >> 5;
  const int lane = threadIdx.x & 31;
  const int lo = lane & 15, hi = lane >> 4;
  const int task  = blockIdx.x * 8 + wv;          // 1024 tasks; one block = one batch slice
  const int b     = task >> 8;
  const int mbase = (task & 255) * 16;

  const half_t* q_b = qt + (size_t)b * N * DQK;
  const half_t* k_b = kt + (size_t)b * N * DQK;
  const half_t* v_b = vm + (size_t)b * C * N;

  // Q A-fragment (held whole loop)
  v16h qfrag;
  {
    const half_t* qr = q_b + (size_t)(mbase + lo) * DQK;
    v8h a0 = *(const v8h*)(qr + 16 * hi);
    v8h a1 = *(const v8h*)(qr + 16 * hi + 8);
#pragma unroll
    for (int h = 0; h < 8; ++h) { qfrag[h] = a0[h]; qfrag[h + 8] = a1[h]; }
  }

  v8f acc[16] = {};
  float mrow[8], lrow[8];
#pragma unroll
  for (int r = 0; r < 8; ++r) { mrow[r] = -1e30f; lrow[r] = 0.0f; }

  // Prologue: stage first V tile
  if (wv == 0) {
    tdm_load_v(v_b, (unsigned int)(uintptr_t)&vtile[0][0]);
    __builtin_amdgcn_s_wait_tensorcnt(0);
  }
  __syncthreads();

  for (int it = 0; it < N / 32; ++it) {
    const int j   = it * 32;
    const int cur = it & 1;

    // Prefetch next V tile into the other buffer (its readers finished last iter)
    if (wv == 0 && j + 32 < N)
      tdm_load_v(v_b + (j + 32), (unsigned int)(uintptr_t)&vtile[cur ^ 1][0]);

    // S = Q K^T for kv columns [j, j+32)
    v16h kf0 = *(const v16h*)(k_b + (size_t)(j + lo) * DQK + 16 * hi);
    v16h kf1 = *(const v16h*)(k_b + (size_t)(j + 16 + lo) * DQK + 16 * hi);
    v8f s0 = {}, s1 = {};
    s0 = WMMA_F16(qfrag, kf0, s0);
    s1 = WMMA_F16(qfrag, kf1, s1);

    float alpha[8], p0[8], p1[8];
#pragma unroll
    for (int r = 0; r < 8; ++r) {
      float mx = fmaxf(s0[r], s1[r]);
      mx = fmaxf(mx, __shfl_xor(mx, 1));
      mx = fmaxf(mx, __shfl_xor(mx, 2));
      mx = fmaxf(mx, __shfl_xor(mx, 4));
      mx = fmaxf(mx, __shfl_xor(mx, 8));
      const float mnew = fmaxf(mrow[r], mx);
      alpha[r] = __expf(mrow[r] - mnew);
      mrow[r]  = mnew;
      p0[r] = __expf(s0[r] - mnew);
      p1[r] = __expf(s1[r] - mnew);
      float sum = p0[r] + p1[r];
      sum += __shfl_xor(sum, 1);
      sum += __shfl_xor(sum, 2);
      sum += __shfl_xor(sum, 4);
      sum += __shfl_xor(sum, 8);
      lrow[r] = lrow[r] * alpha[r] + sum;
    }

    // Relayout P (C/D layout) -> A layout through per-wave LDS slab
#pragma unroll
    for (int r = 0; r < 8; ++r) {
      pbuf[wv][r + 8 * hi][lo]      = (half_t)p0[r];
      pbuf[wv][r + 8 * hi][16 + lo] = (half_t)p1[r];
    }
    asm volatile("s_wait_dscnt 0x0" ::: "memory");
    v16h pfrag;
    {
      const _Float16* pr = &pbuf[wv][lo][0];
      v8h a0 = *(const v8h*)(pr + 16 * hi);
      v8h a1 = *(const v8h*)(pr + 16 * hi + 8);
#pragma unroll
      for (int h = 0; h < 8; ++h) { pfrag[h] = a0[h]; pfrag[h + 8] = a1[h]; }
    }

    // O = alpha*O + P V^T, V fragments read from the TDM-staged LDS tile.
    const _Float16* vt = &vtile[cur][0];
    asm volatile("" : "+v"(vt));   // opaque: TDM wrote this, compiler can't see it
#pragma unroll
    for (int ct = 0; ct < 16; ++ct) {
#pragma unroll
      for (int r = 0; r < 8; ++r) acc[ct][r] *= alpha[r];
      const _Float16* vrow = vt + (size_t)(ct * 16 + lo) * VPAD + 16 * hi;
      v8h b0 = *(const v8h*)(vrow);
      v8h b1 = *(const v8h*)(vrow + 8);
      v16h vf;
#pragma unroll
      for (int h = 0; h < 8; ++h) { vf[h] = b0[h]; vf[h + 8] = b1[h]; }
      acc[ct] = WMMA_F16(pfrag, vf, acc[ct]);
    }

    // Make the prefetched tile visible to the whole block before next iteration
    if (wv == 0 && j + 32 < N) __builtin_amdgcn_s_wait_tensorcnt(0);
    __syncthreads();
  }

  // Epilogue: out = gamma * O/l + x  (32B vector load/store, consecutive in m)
  const float g = gamma_p[0];
  float inv[8];
#pragma unroll
  for (int r = 0; r < 8; ++r) inv[r] = 1.0f / lrow[r];
#pragma unroll
  for (int ct = 0; ct < 16; ++ct) {
    const int c = ct * 16 + lo;
    const size_t off = ((size_t)b * C + c) * N + mbase + 8 * hi;
    v8f xv = *(const v8f*)(x + off);
    v8f o;
#pragma unroll
    for (int r = 0; r < 8; ++r) o[r] = g * acc[ct][r] * inv[r] + xv[r];
    *(v8f*)(out + off) = o;
  }
}

// ---------------------------------------------------------------------------
extern "C" void kernel_launch(void* const* d_in, const int* in_sizes, int n_in,
                              void* d_out, int out_size, void* d_ws, size_t ws_size,
                              hipStream_t stream) {
  const float* x     = (const float*)d_in[0];
  const float* Wq    = (const float*)d_in[1];
  const float* bq    = (const float*)d_in[2];
  const float* Wk    = (const float*)d_in[3];
  const float* bk    = (const float*)d_in[4];
  const float* Wv    = (const float*)d_in[5];
  const float* bv    = (const float*)d_in[6];
  const float* gamma = (const float*)d_in[7];
  float* outp = (float*)d_out;

  // Workspace: Qt (1MB) + Kt (1MB) + V (8MB), all f16
  half_t* qt = (half_t*)d_ws;
  half_t* kt = qt + (size_t)BATCH * N * DQK;
  half_t* vv = kt + (size_t)BATCH * N * DQK;

  dim3 gqk(N / 64, DQK / 16, BATCH);
  proj_qk_kernel<<<gqk, 32, 0, stream>>>(x, Wq, bq, qt);
  proj_qk_kernel<<<gqk, 32, 0, stream>>>(x, Wk, bk, kt);

  dim3 gv(N / 64, C / 16, BATCH);
  proj_v_kernel<<<gv, 32, 0, stream>>>(x, Wv, bv, vv);

  attn_kernel<<<dim3(BATCH * (N / 16) / 8), 256, 0, stream>>>(x, qt, kt, vv, gamma, outp);
}
